// GraphAttentionNetwork_40699110097196
// MI455X (gfx1250) — matrix operand order repaired and checked by reading, not compiled, not run on hardware
//
#include <hip/hip_runtime.h>
#include <hip/hip_bf16.h>

#define N_NODES 50000
#define F_IN    128
#define HEADS   8
#define HID     64
#define C1      (HEADS*HID)            // 512
#define N_EDGES 400000
#define E2      (N_EDGES + N_NODES)    // 450000 (self-loops appended)
#define NEG_SLOPE 0.2f
#define EPS_GAT   1e-16f

typedef __attribute__((ext_vector_type(16))) __bf16 v16bf;
typedef __attribute__((ext_vector_type(8)))  float  v8f;

__device__ __forceinline__ float lrelu(float v) { return v > 0.f ? v : NEG_SLOPE * v; }

__device__ __forceinline__ unsigned short f2bf(float f) {
  __bf16 b = (__bf16)f; unsigned short u; __builtin_memcpy(&u, &b, 2); return u;
}

// Float atomic-max via signed/unsigned integer ordering trick.
// Buffer must be initialized to -inf (0xFF800000).
__device__ __forceinline__ void atomicMaxF(float* addr, float val) {
  if (val >= 0.f) atomicMax((int*)addr, __float_as_int(val));
  else            atomicMin((unsigned int*)addr, __float_as_uint(val));
}

__device__ __forceinline__ void edge_sd(const long long* __restrict__ ei, int e, int& s, int& d) {
  if (e < N_EDGES) { s = (int)ei[e]; d = (int)ei[N_EDGES + e]; }
  else             { s = e - N_EDGES; d = s; }
}

// ---------------------------------------------------------------- K0: W1 -> bf16, TRANSPOSED to [N=512][K=128]
// so each lane's B-fragment (16 consecutive K for one column) is one contiguous 32B vector load.
__global__ void k_convert_w1t(const float* __restrict__ W1, unsigned short* __restrict__ w1t) {
  int i = blockIdx.x * blockDim.x + threadIdx.x;
  if (i >= F_IN * C1) return;
  int col = i >> 7;        // 0..511
  int k   = i & (F_IN - 1);
  w1t[col * F_IN + k] = f2bf(W1[k * C1 + col]);
}

// ---------------------------------------------------------------- K_init: zero/-inf accumulators
__global__ void k_init(float* __restrict__ agg1, float* __restrict__ emax1, float* __restrict__ den1,
                       float* __restrict__ emax2, float* __restrict__ den2, float* __restrict__ agg2) {
  int i = blockIdx.x * blockDim.x + threadIdx.x;
  const float ninf = __int_as_float(0xff800000);
  if (i < N_NODES * C1)    agg1[i] = 0.f;
  if (i < N_NODES * HEADS) { emax1[i] = ninf; den1[i] = 0.f; }
  if (i < N_NODES)         { emax2[i] = ninf; den2[i] = 0.f; agg2[2*i] = 0.f; agg2[2*i+1] = 0.f; }
}

// ---------------------------------------------------------------- K1: h1 = x @ W1 (bf16 WMMA) + alpha epilogue
// grid = N/16 blocks, 256 threads = 8 waves; wave w owns head w (cols w*64 .. w*64+63)
__global__ __launch_bounds__(256) void k_gemm1(
    const float* __restrict__ x, const unsigned short* __restrict__ w1t,
    const float* __restrict__ a_src1, const float* __restrict__ a_dst1,
    float* __restrict__ h1, float* __restrict__ as1, float* __restrict__ ad1) {
  const int lane = threadIdx.x & 31;
  const int w    = threadIdx.x >> 5;   // wave id == head id
  const int hi   = lane >> 4;
  const int n16  = lane & 15;
  const int rowBase = blockIdx.x * 16; // 50000 = 16 * 3125 exactly

  v8f acc[4];
  #pragma unroll
  for (int t = 0; t < 4; ++t)
    #pragma unroll
    for (int j = 0; j < 8; ++j) acc[t][j] = 0.f;

  const float* xr = x + (size_t)(rowBase + n16) * F_IN;

  #pragma unroll
  for (int ks = 0; ks < 4; ++ks) {           // K = 128 = 4 * 32
    // A fragment (16x32 bf16): lane half hi holds K = 16*hi+0..15 (matches B)
    v16bf a;
    const float* ap = xr + 32 * ks + 16 * hi;
    #pragma unroll
    for (int i = 0; i < 8; ++i) a[i]     = (__bf16)ap[i];
    #pragma unroll
    for (int i = 0; i < 8; ++i) a[8 + i] = (__bf16)ap[8 + i];

    #pragma unroll
    for (int t = 0; t < 4; ++t) {
      // B fragment (32x16 bf16): lane holds column n = colBase+n16, elem i = K = 32*ks+16*hi+i.
      // W1t is [col][K] row-major -> 16 contiguous bf16 = one aligned 32-byte vector load.
      const int colBase = w * 64 + 16 * t;
      const v16bf b = *(const v16bf*)(w1t + (size_t)(colBase + n16) * F_IN + 32 * ks + 16 * hi);
      acc[t] = __builtin_amdgcn_wmma_f32_16x16x32_bf16(
          false, a, false, b, (short)0, acc[t], false, false);
    }
  }

  // Epilogue: store h1 tile; fused alpha_src/alpha_dst = h[row,head,:] . a_{src,dst}[head,:]
  float ps[8], pd[8];
  #pragma unroll
  for (int j = 0; j < 8; ++j) { ps[j] = 0.f; pd[j] = 0.f; }

  #pragma unroll
  for (int t = 0; t < 4; ++t) {
    const int col = w * 64 + 16 * t + n16;    // flat col == head*64 + local -> a_src1 index
    const float asv = a_src1[col];
    const float adv = a_dst1[col];
    #pragma unroll
    for (int j = 0; j < 8; ++j) {
      const float v = acc[t][j];              // row = rowBase + j + 8*hi
      h1[(size_t)(rowBase + j + 8 * hi) * C1 + col] = v;
      ps[j] += v * asv;
      pd[j] += v * adv;
    }
  }
  // reduce across the 16 lanes of each half-wave (rows differ only by hi)
  #pragma unroll
  for (int m = 1; m < 16; m <<= 1) {
    #pragma unroll
    for (int j = 0; j < 8; ++j) {
      ps[j] += __shfl_xor(ps[j], m, 32);
      pd[j] += __shfl_xor(pd[j], m, 32);
    }
  }
  if (n16 == 0) {
    #pragma unroll
    for (int j = 0; j < 8; ++j) {
      const int row = rowBase + j + 8 * hi;
      as1[row * HEADS + w] = ps[j];
      ad1[row * HEADS + w] = pd[j];
    }
  }
}

// ---------------------------------------------------------------- K2: per-dst segment max (layer 1)
__global__ void k_edge_max1(const long long* __restrict__ ei,
                            const float* __restrict__ as1, const float* __restrict__ ad1,
                            float* __restrict__ emax1) {
  int e = blockIdx.x * blockDim.x + threadIdx.x;
  if (e >= E2) return;
  int s, d; edge_sd(ei, e, s, d);
  #pragma unroll
  for (int h = 0; h < HEADS; ++h) {
    float v = lrelu(as1[s * HEADS + h] + ad1[d * HEADS + h]);
    atomicMaxF(emax1 + d * HEADS + h, v);
  }
}

// ---------------------------------------------------------------- K3: unnormalized accumulate (layer 1)
// one wave per edge: lanes 0-7 compute per-head exp weights, broadcast, 512-ch scatter-add
__global__ __launch_bounds__(256) void k_edge_accum1(
    const long long* __restrict__ ei,
    const float* __restrict__ as1, const float* __restrict__ ad1,
    const float* __restrict__ emax1, const float* __restrict__ h1,
    float* __restrict__ den1, float* __restrict__ agg1) {
  const int lane = threadIdx.x & 31;
  const int e = blockIdx.x * 8 + (threadIdx.x >> 5);
  if (e >= E2) return;
  int s, d; edge_sd(ei, e, s, d);

  const float* hs = h1 + (size_t)s * C1;
  __builtin_prefetch(hs, 0, 1);              // global_prefetch_b8: warm the gathered row

  float ww = 0.f;
  if (lane < HEADS) {
    float v = lrelu(as1[s * HEADS + lane] + ad1[d * HEADS + lane]);
    ww = __expf(v - emax1[d * HEADS + lane]);
    atomicAdd(den1 + d * HEADS + lane, ww);
  }
  float* ag = agg1 + (size_t)d * C1;
  #pragma unroll
  for (int k = 0; k < 16; ++k) {             // c = 32k + lane; head(c) = k>>1 (uniform per k)
    const float wk = __shfl(ww, k >> 1, 32);
    const int c = k * 32 + lane;
    atomicAdd(ag + c, wk * hs[c]);
  }
}

// ---------------------------------------------------------------- K4: normalize + bias + ReLU (in place)
__global__ void k_final1(float* __restrict__ agg1, const float* __restrict__ den1,
                         const float* __restrict__ b1) {
  int i = blockIdx.x * blockDim.x + threadIdx.x;
  if (i >= N_NODES * C1) return;
  int n = i >> 9;
  int h = (i >> 6) & 7;
  float v = agg1[i] / (den1[n * HEADS + h] + EPS_GAT) + b1[i & (C1 - 1)];
  agg1[i] = v > 0.f ? v : 0.f;
}

// ---------------------------------------------------------------- K5: layer-2 linear (512 -> 2) + alphas
__global__ __launch_bounds__(256) void k_linear2(
    const float* __restrict__ z, const float* __restrict__ W2,
    const float* __restrict__ a_src2, const float* __restrict__ a_dst2,
    float* __restrict__ h2, float* __restrict__ as2, float* __restrict__ ad2) {
  const int lane = threadIdx.x & 31;
  const int n = blockIdx.x * 8 + (threadIdx.x >> 5);
  if (n >= N_NODES) return;
  const float* zr = z + (size_t)n * C1;
  float s0 = 0.f, s1 = 0.f;
  #pragma unroll
  for (int k = 0; k < 16; ++k) {
    const int c = k * 32 + lane;
    const float v = zr[c];
    s0 += v * W2[2 * c];
    s1 += v * W2[2 * c + 1];
  }
  #pragma unroll
  for (int m = 16; m >= 1; m >>= 1) {
    s0 += __shfl_xor(s0, m, 32);
    s1 += __shfl_xor(s1, m, 32);
  }
  if (lane == 0) {
    h2[2 * n] = s0; h2[2 * n + 1] = s1;
    as2[n] = s0 * a_src2[0] + s1 * a_src2[1];
    ad2[n] = s0 * a_dst2[0] + s1 * a_dst2[1];
  }
}

// ---------------------------------------------------------------- K6/K7: layer-2 edge passes
__global__ void k_edge_max2(const long long* __restrict__ ei,
                            const float* __restrict__ as2, const float* __restrict__ ad2,
                            float* __restrict__ emax2) {
  int e = blockIdx.x * blockDim.x + threadIdx.x;
  if (e >= E2) return;
  int s, d; edge_sd(ei, e, s, d);
  atomicMaxF(emax2 + d, lrelu(as2[s] + ad2[d]));
}

__global__ void k_edge_accum2(const long long* __restrict__ ei,
                              const float* __restrict__ as2, const float* __restrict__ ad2,
                              const float* __restrict__ emax2, const float* __restrict__ h2,
                              float* __restrict__ den2, float* __restrict__ agg2) {
  int e = blockIdx.x * blockDim.x + threadIdx.x;
  if (e >= E2) return;
  int s, d; edge_sd(ei, e, s, d);
  float w = __expf(lrelu(as2[s] + ad2[d]) - emax2[d]);
  atomicAdd(den2 + d, w);
  atomicAdd(agg2 + 2 * d,     w * h2[2 * s]);
  atomicAdd(agg2 + 2 * d + 1, w * h2[2 * s + 1]);
}

// ---------------------------------------------------------------- K8: final output
__global__ void k_final2(const float* __restrict__ agg2, const float* __restrict__ den2,
                         const float* __restrict__ b2, float* __restrict__ out) {
  int n = blockIdx.x * blockDim.x + threadIdx.x;
  if (n >= N_NODES) return;
  float dn = den2[n] + EPS_GAT;
  out[2 * n]     = agg2[2 * n]     / dn + b2[0];
  out[2 * n + 1] = agg2[2 * n + 1] / dn + b2[1];
}

// ================================================================ launch
extern "C" void kernel_launch(void* const* d_in, const int* in_sizes, int n_in,
                              void* d_out, int out_size, void* d_ws, size_t ws_size,
                              hipStream_t stream) {
  (void)in_sizes; (void)n_in; (void)out_size; (void)ws_size;
  const float*     x      = (const float*)d_in[0];
  const long long* ei     = (const long long*)d_in[1];   // int64 [2, E]
  const float*     W1     = (const float*)d_in[2];
  const float*     a_src1 = (const float*)d_in[3];
  const float*     a_dst1 = (const float*)d_in[4];
  const float*     b1     = (const float*)d_in[5];
  const float*     W2     = (const float*)d_in[6];
  const float*     a_src2 = (const float*)d_in[7];
  const float*     a_dst2 = (const float*)d_in[8];
  const float*     b2     = (const float*)d_in[9];
  float*           out    = (float*)d_out;

  char* p = (char*)d_ws;
  auto carve = [&](size_t bytes) -> char* {
    char* q = p; p += (bytes + 255) & ~(size_t)255; return q;
  };
  float*          h1    = (float*)carve((size_t)N_NODES * C1 * 4);      // 102.4 MB
  float*          agg1  = (float*)carve((size_t)N_NODES * C1 * 4);      // 102.4 MB (reused as layer-2 input)
  float*          as1   = (float*)carve((size_t)N_NODES * HEADS * 4);
  float*          ad1   = (float*)carve((size_t)N_NODES * HEADS * 4);
  float*          emax1 = (float*)carve((size_t)N_NODES * HEADS * 4);
  float*          den1  = (float*)carve((size_t)N_NODES * HEADS * 4);
  unsigned short* w1t   = (unsigned short*)carve((size_t)F_IN * C1 * 2); // transposed bf16 W1
  float*          h2    = (float*)carve((size_t)N_NODES * 2 * 4);
  float*          as2   = (float*)carve((size_t)N_NODES * 4);
  float*          ad2   = (float*)carve((size_t)N_NODES * 4);
  float*          emax2 = (float*)carve((size_t)N_NODES * 4);
  float*          den2  = (float*)carve((size_t)N_NODES * 4);
  float*          agg2  = (float*)carve((size_t)N_NODES * 2 * 4);

  const int B = 256;
  hipLaunchKernelGGL(k_convert_w1t, dim3((F_IN * C1 + B - 1) / B), dim3(B), 0, stream, W1, w1t);
  hipLaunchKernelGGL(k_init, dim3((N_NODES * C1 + B - 1) / B), dim3(B), 0, stream,
                     agg1, emax1, den1, emax2, den2, agg2);
  hipLaunchKernelGGL(k_gemm1, dim3(N_NODES / 16), dim3(B), 0, stream,
                     x, w1t, a_src1, a_dst1, h1, as1, ad1);
  hipLaunchKernelGGL(k_edge_max1, dim3((E2 + B - 1) / B), dim3(B), 0, stream, ei, as1, ad1, emax1);
  hipLaunchKernelGGL(k_edge_accum1, dim3((E2 + 7) / 8), dim3(B), 0, stream,
                     ei, as1, ad1, emax1, h1, den1, agg1);
  hipLaunchKernelGGL(k_final1, dim3((N_NODES * C1 + B - 1) / B), dim3(B), 0, stream, agg1, den1, b1);
  hipLaunchKernelGGL(k_linear2, dim3((N_NODES + 7) / 8), dim3(B), 0, stream,
                     agg1, W2, a_src2, a_dst2, h2, as2, ad2);
  hipLaunchKernelGGL(k_edge_max2, dim3((E2 + B - 1) / B), dim3(B), 0, stream, ei, as2, ad2, emax2);
  hipLaunchKernelGGL(k_edge_accum2, dim3((E2 + B - 1) / B), dim3(B), 0, stream,
                     ei, as2, ad2, emax2, h2, den2, agg2);
  hipLaunchKernelGGL(k_final2, dim3((N_NODES + B - 1) / B), dim3(B), 0, stream, agg2, den2, b2, out);
}
